// LlamaAttention_47639777247389
// MI455X (gfx1250) — matrix operand order repaired and checked by reading, not compiled, not run on hardware
//
#include <hip/hip_runtime.h>
#include <hip/hip_bf16.h>

// ---------------------------------------------------------------------------
// Llama attention block for MI455X (gfx1250, wave32, WMMA 16x16x32 f16->f32)
// ---------------------------------------------------------------------------

typedef _Float16 f16;
typedef __attribute__((ext_vector_type(16))) _Float16 v16h;
typedef __attribute__((ext_vector_type(8)))  _Float16 v8h;
typedef __attribute__((ext_vector_type(8)))  float    v8f;
typedef __attribute__((ext_vector_type(4)))  int      v4i;

union frag16 { v16h v; v8h h[2]; };

static constexpr int Bq  = 2;
static constexpr int Sq  = 2048;
static constexpr int Hq  = 4096;
static constexpr int NHq = 32;
static constexpr int NKVq= 8;
static constexpr int Dq  = 128;

static __device__ __forceinline__ v8f wmma_f16(v16h a, v16h b, v8f c) {
  return __builtin_amdgcn_wmma_f32_16x16x32_f16(
      /*neg_a=*/false, a, /*neg_b=*/false, b,
      /*c_mod=*/(short)0, c, /*reuse_a=*/false, /*reuse_b=*/false);
}

// Optional gfx1250 async global->LDS staging (ASYNCcnt-tracked), if the
// toolchain exposes the builtins; otherwise fall back to plain loads.
#if defined(__has_builtin)
#if __has_builtin(__builtin_amdgcn_global_load_async_to_lds_b128) && \
    __has_builtin(__builtin_amdgcn_s_wait_asynccnt)
#define USE_ASYNC_LDS 1
#endif
#endif

#ifdef USE_ASYNC_LDS
typedef __attribute__((address_space(1))) v4i* g4p;   // global int4*
typedef __attribute__((address_space(3))) v4i* l4p;   // LDS int4*
static __device__ __forceinline__ void async_cp16(const f16* g, f16* l) {
  __builtin_amdgcn_global_load_async_to_lds_b128(
      (g4p)(void*)const_cast<f16*>(g), (l4p)(void*)l, 0, 0);
}
#endif

// --------------------------- elementwise converts ---------------------------

__global__ void conv_f16_kernel(const float* __restrict__ x,
                                f16* __restrict__ y, size_t n) {
  size_t i = (size_t)blockIdx.x * blockDim.x + threadIdx.x;
  if (i < n) y[i] = (f16)x[i];
}

// B[k][n] (KxN row-major, f32) -> BT[n][k] (NxK row-major, f16)
__global__ void conv_bt_kernel(const float* __restrict__ Bm,
                               f16* __restrict__ BT, int K, int N) {
  size_t i = (size_t)blockIdx.x * blockDim.x + threadIdx.x;
  size_t total = (size_t)K * N;
  if (i >= total) return;
  int k = (int)(i % K);
  int n = (int)(i / K);
  BT[i] = (f16)Bm[(size_t)k * N + n];
}

// X: [B*S, nheads*D] f32 -> Y: [B][nheads][S][D] f16, with RoPE applied.
__global__ void rope_convert_kernel(const float* __restrict__ X,
                                    const float* __restrict__ cosT,
                                    const float* __restrict__ sinT,
                                    f16* __restrict__ Y,
                                    int nheads, float scale) {
  size_t i = (size_t)blockIdx.x * blockDim.x + threadIdx.x;
  size_t total = (size_t)Bq * Sq * nheads * Dq;
  if (i >= total) return;
  int d = (int)(i % Dq);   size_t t = i / Dq;
  int s = (int)(t % Sq);   t /= Sq;
  int h = (int)(t % nheads);
  int b = (int)(t / nheads);
  const float* xp = X + ((size_t)b * Sq + s) * (size_t)(nheads * Dq) + (size_t)h * Dq;
  float v   = xp[d];
  float rot = (d < Dq / 2) ? -xp[d + Dq / 2] : xp[d - Dq / 2];
  float c  = cosT[(size_t)s * Dq + d];
  float sn = sinT[(size_t)s * Dq + d];
  Y[i] = (f16)((v * c + rot * sn) * scale);
}

// V: [B*S, NKV*D] f32 -> VT: [B][NKV][D][S] f16
__global__ void conv_vt_kernel(const float* __restrict__ V, f16* __restrict__ VT) {
  size_t i = (size_t)blockIdx.x * blockDim.x + threadIdx.x;
  size_t total = (size_t)Bq * NKVq * Dq * Sq;
  if (i >= total) return;
  int s = (int)(i % Sq);   size_t t = i / Sq;
  int d = (int)(t % Dq);   t /= Dq;
  int h = (int)(t % NKVq);
  int b = (int)(t / NKVq);
  VT[i] = (f16)V[((size_t)b * Sq + s) * (size_t)(NKVq * Dq) + (size_t)h * Dq + d];
}

// ------------------------------ WMMA GEMM ----------------------------------
// C[M,N] f32 = A[M,K] f16 (row-major) x BT[N,K] f16 (B transposed, row-major).
// Block tile 128x128, 8 waves as 4(M)x2(N); each wave 32x64 = 2x4 WMMA tiles.
__global__ __launch_bounds__(256) void gemm_wmma_kernel(
    const f16* __restrict__ A, const f16* __restrict__ BT,
    float* __restrict__ C, int M, int N, int K) {
  __shared__ __align__(16) f16 As[128][32];
  __shared__ __align__(16) f16 Bs[128][32];

  const int t    = threadIdx.x;
  const int lane = t & 31;
  const int w    = t >> 5;
  const int wm   = w & 3;               // 0..3 -> M offset wm*32
  const int wn   = w >> 2;              // 0..1 -> N offset wn*64
  const int lrow = lane & 15;
  const int kh   = lane >> 4;           // which half of the wave
  const int bm   = blockIdx.x * 128;
  const int bn   = blockIdx.y * 128;

  v8f acc[2][4] = {};

  for (int k0 = 0; k0 < K; k0 += 32) {
    // stage A and BT tiles: each 128x32 f16 = 512 chunks of 8 f16
    #pragma unroll
    for (int i = 0; i < 2; ++i) {
      int cid = t + i * 256;
      int r   = cid >> 2;
      int c8  = (cid & 3) * 8;
#ifdef USE_ASYNC_LDS
      async_cp16(&A [(size_t)(bm + r) * K + k0 + c8], &As[r][c8]);
      async_cp16(&BT[(size_t)(bn + r) * K + k0 + c8], &Bs[r][c8]);
#else
      *(v8h*)&As[r][c8] = *(const v8h*)&A [(size_t)(bm + r) * K + k0 + c8];
      *(v8h*)&Bs[r][c8] = *(const v8h*)&BT[(size_t)(bn + r) * K + k0 + c8];
#endif
    }
    if (k0 + 32 < K) {  // prefetch next A strip (global_prefetch_b8)
      __builtin_prefetch(&A[(size_t)(bm + (t >> 1)) * K + k0 + 32], 0, 1);
    }
#ifdef USE_ASYNC_LDS
    __builtin_amdgcn_s_wait_asynccnt(0);
#endif
    __syncthreads();

    frag16 a[2];  // A layout: lanes 0-15 K=0..7,16..23 ; lanes 16-31 K=8..15,24..31
    #pragma unroll
    for (int af = 0; af < 2; ++af) {
      a[af].h[0] = *(const v8h*)&As[wm * 32 + af * 16 + lrow][kh * 8];
      a[af].h[1] = *(const v8h*)&As[wm * 32 + af * 16 + lrow][16 + kh * 8];
    }
    #pragma unroll
    for (int nt = 0; nt < 4; ++nt) {
      frag16 b;  // B layout: lanes 0-15 K=0..7,16..23 ; lanes 16-31 K=8..15,24..31
      b.h[0] = *(const v8h*)&Bs[wn * 64 + nt * 16 + lrow][kh * 8];
      b.h[1] = *(const v8h*)&Bs[wn * 64 + nt * 16 + lrow][16 + kh * 8];
      #pragma unroll
      for (int af = 0; af < 2; ++af)
        acc[af][nt] = wmma_f16(a[af].v, b.v, acc[af][nt]);
    }
    __syncthreads();
  }

  #pragma unroll
  for (int af = 0; af < 2; ++af)
    #pragma unroll
    for (int nt = 0; nt < 4; ++nt)
      #pragma unroll
      for (int r = 0; r < 8; ++r) {
        int row = bm + wm * 32 + af * 16 + r + kh * 8;  // C: VGPR r -> M=r / r+8
        int col = bn + wn * 64 + nt * 16 + lrow;
        C[(size_t)row * N + col] = acc[af][nt][r];
      }
}

// --------------------------- flash attention -------------------------------
// Q: [B][NH][S][D] f16 (scale folded in), K: [B][NKV][S][D] f16,
// VT: [B][NKV][D][S] f16.  Out: [B*S][NH*D] f16 (A-operand for out-proj).
// Block = 8 waves; wave w owns 16 query rows of a 128-row q-tile.
__global__ __launch_bounds__(256) void flash_attn_kernel(
    const f16* __restrict__ Q, const f16* __restrict__ Kc,
    const f16* __restrict__ VT, f16* __restrict__ O16) {
  __shared__ __align__(16) f16 Plds[8][16][32];

  const int t    = threadIdx.x;
  const int lane = t & 31;
  const int w    = t >> 5;
  const int lrow = lane & 15;
  const int kh   = lane >> 4;
  const int bh   = blockIdx.x;           // b*NH + h
  const int qt   = blockIdx.y;           // q tile of 128 rows
  const int b    = bh / NHq;
  const int h    = bh % NHq;
  const int hk   = h / (NHq / NKVq);

  const f16* qp = Q  + (((size_t)b * NHq  + h ) * Sq + (size_t)qt * 128 + w * 16) * Dq;
  const f16* kp = Kc + (((size_t)b * NKVq + hk) * Sq) * (size_t)Dq;
  const f16* vp = VT + (((size_t)b * NKVq + hk) * Dq) * (size_t)Sq;

  // Q fragments for this wave's 16 rows, all of D=128 (4 k-steps of 32)
  frag16 qa[4];
  #pragma unroll
  for (int ks = 0; ks < 4; ++ks) {
    qa[ks].h[0] = *(const v8h*)&qp[(size_t)lrow * Dq + ks * 32 + kh * 8];
    qa[ks].h[1] = *(const v8h*)&qp[(size_t)lrow * Dq + ks * 32 + kh * 8 + 16];
  }

  float m[8], l[8];
  v8f   o[8] = {};
  #pragma unroll
  for (int r = 0; r < 8; ++r) { m[r] = -1e30f; l[r] = 0.0f; }

  const int qrow_lo = qt * 128 + w * 16;   // query row of accumulator r=0, kh=0
  const int nkt     = (qt + 1) * 4;        // causal: 32-key tiles to cover

  for (int kt = 0; kt < nkt; ++kt) {
    const int key_base = kt * 32;

    // ---- scores: S(16x32) = Q(16x128) @ K^T, two 16-key halves ----
    v8f st[2] = {};
    #pragma unroll
    for (int half = 0; half < 2; ++half) {
      const f16* kr = kp + (size_t)(key_base + half * 16 + lrow) * Dq;
      #pragma unroll
      for (int ks = 0; ks < 4; ++ks) {
        frag16 bf;
        bf.h[0] = *(const v8h*)&kr[ks * 32 + kh * 8];
        bf.h[1] = *(const v8h*)&kr[ks * 32 + kh * 8 + 16];
        st[half] = wmma_f16(qa[ks].v, bf.v, st[half]);
      }
    }

    // ---- causal mask ----
    #pragma unroll
    for (int half = 0; half < 2; ++half)
      #pragma unroll
      for (int r = 0; r < 8; ++r) {
        int key = key_base + half * 16 + lrow;
        int qr  = qrow_lo + r + kh * 8;
        if (key > qr) st[half][r] = -1e9f;
      }

    // ---- online softmax (row reductions across 16-lane halves) ----
    float alpha[8];
    #pragma unroll
    for (int r = 0; r < 8; ++r) {
      float mx = fmaxf(st[0][r], st[1][r]);
      #pragma unroll
      for (int msk = 8; msk >= 1; msk >>= 1)
        mx = fmaxf(mx, __shfl_xor(mx, msk, 32));
      float mn = fmaxf(m[r], mx);
      alpha[r] = __expf(m[r] - mn);
      m[r] = mn;
      float p0 = __expf(st[0][r] - mn);
      float p1 = __expf(st[1][r] - mn);
      st[0][r] = p0;
      st[1][r] = p1;
      float rs = p0 + p1;
      #pragma unroll
      for (int msk = 8; msk >= 1; msk >>= 1)
        rs += __shfl_xor(rs, msk, 32);
      l[r] = l[r] * alpha[r] + rs;
    }
    #pragma unroll
    for (int nt = 0; nt < 8; ++nt)
      #pragma unroll
      for (int r = 0; r < 8; ++r) o[nt][r] *= alpha[r];

    // ---- reshape P: accumulator (C) layout -> A-fragment layout via LDS ----
    #pragma unroll
    for (int half = 0; half < 2; ++half)
      #pragma unroll
      for (int r = 0; r < 8; ++r)
        Plds[w][r + kh * 8][half * 16 + lrow] = (f16)st[half][r];
    __syncthreads();
    frag16 pa;
    pa.h[0] = *(const v8h*)&Plds[w][lrow][kh * 8];
    pa.h[1] = *(const v8h*)&Plds[w][lrow][16 + kh * 8];
    __syncthreads();

    // ---- O += P(16x32) @ V(32x128), 8 d-tiles ----
    #pragma unroll
    for (int nt = 0; nt < 8; ++nt) {
      const f16* vr = vp + (size_t)(nt * 16 + lrow) * Sq + key_base;
      frag16 vb;
      vb.h[0] = *(const v8h*)&vr[kh * 8];
      vb.h[1] = *(const v8h*)&vr[kh * 8 + 16];
      o[nt] = wmma_f16(pa.v, vb.v, o[nt]);
    }
  }

  // ---- normalize and store as f16 in [B*S][NH*D] (token-major) ----
  #pragma unroll
  for (int nt = 0; nt < 8; ++nt)
    #pragma unroll
    for (int r = 0; r < 8; ++r) {
      int qr = qrow_lo + r + kh * 8;
      int d  = nt * 16 + lrow;
      float val = o[nt][r] / l[r];
      O16[(((size_t)b * Sq + qr) * NHq + h) * Dq + d] = (f16)val;
    }
}

// ------------------------------- launcher ----------------------------------

extern "C" void kernel_launch(void* const* d_in, const int* in_sizes, int n_in,
                              void* d_out, int out_size, void* d_ws, size_t ws_size,
                              hipStream_t stream) {
  const float* hs   = (const float*)d_in[0];
  const float* wq   = (const float*)d_in[1];
  const float* wk   = (const float*)d_in[2];
  const float* wv   = (const float*)d_in[3];
  const float* wo   = (const float*)d_in[4];
  const float* cosT = (const float*)d_in[5];
  const float* sinT = (const float*)d_in[6];
  // d_in[7] = attn_mask (causal) handled analytically.
  float* out = (float*)d_out;

  const size_t BS   = (size_t)Bq * Sq;          // 4096 tokens
  const size_t NHD  = (size_t)NHq * Dq;         // 4096
  const size_t NKVD = (size_t)NKVq * Dq;        // 1024

  // workspace layout
  char* ws = (char*)d_ws;
  size_t off = 0;
  auto take = [&](size_t bytes) -> char* {
    char* p = ws + off;
    off = (off + bytes + 255) & ~(size_t)255;
    return p;
  };
  f16*   hs16  = (f16*)  take(BS * Hq   * sizeof(f16));
  f16*   wqT   = (f16*)  take(NHD  * Hq * sizeof(f16));   // [N=NHD][K=H]
  f16*   wkT   = (f16*)  take(NKVD * Hq * sizeof(f16));
  f16*   wvT   = (f16*)  take(NKVD * Hq * sizeof(f16));
  f16*   woT   = (f16*)  take((size_t)Hq * NHD * sizeof(f16)); // [N=H][K=NHD]
  float* qf32  = (float*)take(BS * NHD  * sizeof(float));
  float* kf32  = (float*)take(BS * NKVD * sizeof(float));
  float* vf32  = (float*)take(BS * NKVD * sizeof(float));
  f16*   q16   = (f16*)  take(BS * NHD  * sizeof(f16));
  f16*   k16   = (f16*)  take(BS * NKVD * sizeof(f16));
  f16*   vt16  = (f16*)  take(BS * NKVD * sizeof(f16));
  f16*   attn16= (f16*)  take(BS * NHD  * sizeof(f16));
  (void)ws_size; (void)in_sizes; (void)n_in; (void)out_size;

  const int TB = 256;
  auto blocks1d = [](size_t n, int tb) { return (unsigned)((n + tb - 1) / tb); };

  // 1) operand conversion
  conv_f16_kernel<<<blocks1d(BS * Hq, TB), TB, 0, stream>>>(hs, hs16, BS * Hq);
  conv_bt_kernel<<<blocks1d((size_t)Hq * NHD,  TB), TB, 0, stream>>>(wq, wqT, Hq, (int)NHD);
  conv_bt_kernel<<<blocks1d((size_t)Hq * NKVD, TB), TB, 0, stream>>>(wk, wkT, Hq, (int)NKVD);
  conv_bt_kernel<<<blocks1d((size_t)Hq * NKVD, TB), TB, 0, stream>>>(wv, wvT, Hq, (int)NKVD);
  conv_bt_kernel<<<blocks1d((size_t)NHD * Hq,  TB), TB, 0, stream>>>(wo, woT, (int)NHD, Hq);

  // 2) QKV projections (WMMA GEMMs), block tile 128x128
  gemm_wmma_kernel<<<dim3((unsigned)(BS / 128), (unsigned)(NHD / 128)),  TB, 0, stream>>>(
      hs16, wqT, qf32, (int)BS, (int)NHD, Hq);
  gemm_wmma_kernel<<<dim3((unsigned)(BS / 128), (unsigned)(NKVD / 128)), TB, 0, stream>>>(
      hs16, wkT, kf32, (int)BS, (int)NKVD, Hq);
  gemm_wmma_kernel<<<dim3((unsigned)(BS / 128), (unsigned)(NKVD / 128)), TB, 0, stream>>>(
      hs16, wvT, vf32, (int)BS, (int)NKVD, Hq);

  // 3) RoPE + layout conversion (fold 1/sqrt(D) into Q)
  const float scale = 0.088388347648318447f;   // 1/sqrt(128)
  rope_convert_kernel<<<blocks1d(BS * NHD,  TB), TB, 0, stream>>>(qf32, cosT, sinT, q16, NHq,  scale);
  rope_convert_kernel<<<blocks1d(BS * NKVD, TB), TB, 0, stream>>>(kf32, cosT, sinT, k16, NKVq, 1.0f);
  conv_vt_kernel<<<blocks1d(BS * NKVD, TB), TB, 0, stream>>>(vf32, vt16);

  // 4) flash attention (WMMA)
  flash_attn_kernel<<<dim3((unsigned)(Bq * NHq), (unsigned)(Sq / 128)), TB, 0, stream>>>(
      q16, k16, vt16, attn16);

  // 5) output projection straight into d_out (fp32)
  gemm_wmma_kernel<<<dim3((unsigned)(BS / 128), (unsigned)(Hq / 128)), TB, 0, stream>>>(
      attn16, woT, out, (int)BS, Hq, (int)NHD);
}
